// MultiHeadTemporalAttention__25056839204907
// MI455X (gfx1250) — matrix operand order repaired and checked by reading, not compiled, not run on hardware
//
#include <hip/hip_runtime.h>
#include <hip/hip_bf16.h>
#include <math.h>

#define B_  4
#define T_  1024
#define D_  1024
#define H_  16
#define HD_ 64
#define EPS_ 1e-5f

typedef __bf16 bf16;
typedef bf16  v16bf __attribute__((ext_vector_type(16)));
typedef bf16  v8bf  __attribute__((ext_vector_type(8)));
typedef bf16  v4bf  __attribute__((ext_vector_type(4)));
typedef float v8f   __attribute__((ext_vector_type(8)));
typedef float v4f   __attribute__((ext_vector_type(4)));

static __device__ __forceinline__ v8f vzero8() {
  v8f z;
#pragma unroll
  for (int i = 0; i < 8; ++i) z[i] = 0.0f;
  return z;
}

static __device__ __forceinline__ v8f wmma_bf16(v16bf a, v16bf b, v8f c) {
  // D = A(16x32 bf16) x B(32x16 bf16) + C(16x16 f32)
  return __builtin_amdgcn_wmma_f32_16x16x32_bf16(false, a, false, b, (short)0, c,
                                                 false, false);
}

// ---- Fragment loaders (ISA 7.12.2 layouts, wave32) --------------------------
// A (16x32 MxK): lane row M = lane&15; h = lane>>4; element e -> K =
// ((e>>3)<<4) + h*8 + (e&7)  => two contiguous 16B chunks.
static __device__ __forceinline__ v16bf a_frag_bf16(const bf16* __restrict__ row,
                                                    int k0, int h) {
  v8bf lo = *(const v8bf*)(row + k0 + h * 8);
  v8bf hi = *(const v8bf*)(row + k0 + 16 + h * 8);
  v16bf r;
#pragma unroll
  for (int i = 0; i < 8; ++i) { r[i] = lo[i]; r[8 + i] = hi[i]; }
  return r;
}

// A-fragment built from f32 source (attn read-back path only).
static __device__ __forceinline__ v16bf a_frag_f32(const float* __restrict__ row,
                                                   int k0, int h) {
  const float* p0 = row + k0 + h * 8;
  const float* p1 = row + k0 + 16 + h * 8;
  v4f x0 = *(const v4f*)p0;
  v4f x1 = *(const v4f*)(p0 + 4);
  v4f y0 = *(const v4f*)p1;
  v4f y1 = *(const v4f*)(p1 + 4);
  v16bf r;
#pragma unroll
  for (int i = 0; i < 4; ++i) {
    r[i]      = (bf16)x0[i];
    r[4 + i]  = (bf16)x1[i];
    r[8 + i]  = (bf16)y0[i];
    r[12 + i] = (bf16)y1[i];
  }
  return r;
}

// B (32x16 KxN): lane column N = lane&15; element e -> K = k0 +
// ((e>>3)<<4) + h*8 + (e&7) (same as A). Two aligned 16B loads
// (rowN is the contiguous memory row storing column N).
static __device__ __forceinline__ v16bf b_frag_bf16(const bf16* __restrict__ rowN,
                                                    int k0, int h) {
  v8bf lo = *(const v8bf*)(rowN + k0 + h * 8);
  v8bf hi = *(const v8bf*)(rowN + k0 + 16 + h * 8);
  v16bf r;
#pragma unroll
  for (int i = 0; i < 8; ++i) { r[i] = lo[i]; r[8 + i] = hi[i]; }
  return r;
}

// ---- Kernel 0: f32 -> bf16 pre-conversion (x and weights) -------------------
__global__ __launch_bounds__(256)
void f32_to_bf16_kernel(const float* __restrict__ in, bf16* __restrict__ out,
                        int n4) {
  const int i = blockIdx.x * 256 + threadIdx.x;
  if (i < n4) {
    const v4f v = ((const v4f*)in)[i];
    v4bf o;
#pragma unroll
    for (int k = 0; k < 4; ++k) o[k] = (bf16)v[k];
    ((v4bf*)out)[i] = o;
  }
}

// ---- Kernel 1: QKV projections: C[4096,1024] = Xbf @ Wbf^T + b, bf16 out ----
// Wave tile 32(M) x 64(N): 2 A-frags, 4 B-frags, 8 wmma per K-step.
// mode 0: out[((b*H+h)*T + t)*HD + d]   (Q and K, row-major per head)
// mode 1: out[((b*H+h)*HD + d)*T + t]   (V stored transposed for ctx B-frags)
__global__ __launch_bounds__(128, 1)
void qkv_proj_kernel(const bf16* __restrict__ X, const bf16* __restrict__ W,
                     const float* __restrict__ bias, bf16* __restrict__ out,
                     int mode) {
  const int lane = threadIdx.x & 31;
  const int wave = threadIdx.x >> 5;
  const int r = lane & 15, h = lane >> 4;
  const int m0 = blockIdx.y * 128 + wave * 32;
  const int n0 = blockIdx.x * 64;
  const bf16* arow0 = X + (size_t)(m0 + r) * D_;
  const bf16* arow1 = X + (size_t)(m0 + 16 + r) * D_;
  const bf16* wrow[4];
#pragma unroll
  for (int j = 0; j < 4; ++j) wrow[j] = W + (size_t)(n0 + 16 * j + r) * D_;

  v8f acc[2][4];
#pragma unroll
  for (int i = 0; i < 2; ++i)
#pragma unroll
    for (int j = 0; j < 4; ++j) acc[i][j] = vzero8();

  for (int k0 = 0; k0 < D_; k0 += 32) {
    const v16bf a0 = a_frag_bf16(arow0, k0, h);
    const v16bf a1 = a_frag_bf16(arow1, k0, h);
    const v16bf b0 = b_frag_bf16(wrow[0], k0, h);
    const v16bf b1 = b_frag_bf16(wrow[1], k0, h);
    const v16bf b2 = b_frag_bf16(wrow[2], k0, h);
    const v16bf b3 = b_frag_bf16(wrow[3], k0, h);
    acc[0][0] = wmma_bf16(a0, b0, acc[0][0]);
    acc[0][1] = wmma_bf16(a0, b1, acc[0][1]);
    acc[0][2] = wmma_bf16(a0, b2, acc[0][2]);
    acc[0][3] = wmma_bf16(a0, b3, acc[0][3]);
    acc[1][0] = wmma_bf16(a1, b0, acc[1][0]);
    acc[1][1] = wmma_bf16(a1, b1, acc[1][1]);
    acc[1][2] = wmma_bf16(a1, b2, acc[1][2]);
    acc[1][3] = wmma_bf16(a1, b3, acc[1][3]);
  }

#pragma unroll
  for (int j = 0; j < 4; ++j) {
    const int n = n0 + 16 * j + r;
    const int hh = n >> 6, d = n & 63;
    const float bv = bias[n];
#pragma unroll
    for (int i = 0; i < 2; ++i)
#pragma unroll
      for (int v = 0; v < 8; ++v) {
        const int m = m0 + 16 * i + v + 8 * h;  // C layout: row = v + 8*(lane>>4)
        const int bb = m >> 10, t = m & 1023;
        const float val = acc[i][j][v] + bv;
        const size_t idx = (mode == 0)
            ? ((size_t)((bb * H_ + hh) * T_ + t) * HD_ + d)
            : ((size_t)((bb * H_ + hh) * HD_ + d) * T_ + t);
        out[idx] = (bf16)val;
      }
  }
}

// ---- Kernel 2: scores + softmax, fully in-register per wave -----------------
// One wave owns a 16x1024 strip of scores for one (b,h): 64 v8f accumulators
// (512 VGPRs/lane; legal at 1 wave/SIMD under wave32's 1024-VGPR budget).
__global__ __launch_bounds__(128, 1)
void attn_scores_softmax_kernel(const bf16* __restrict__ Q, const bf16* __restrict__ K,
                                const float* __restrict__ mask,
                                float* __restrict__ attn) {
  const int lane = threadIdx.x & 31;
  const int wave = threadIdx.x >> 5;
  const int r = lane & 15, h = lane >> 4;
  const int bh = blockIdx.y;
  const int m0 = (blockIdx.x * 4 + wave) * 16;
  const bf16* Qb = Q + (size_t)bh * T_ * HD_;
  const bf16* Kb = K + (size_t)bh * T_ * HD_;
  const bf16* qrow = Qb + (size_t)(m0 + r) * HD_;
  const v16bf qa0 = a_frag_bf16(qrow, 0, h);
  const v16bf qa1 = a_frag_bf16(qrow, 32, h);

  v8f acc[64];
#pragma unroll
  for (int j = 0; j < 64; ++j) acc[j] = vzero8();

#pragma unroll
  for (int j = 0; j < 64; ++j) {
    const bf16* krow = Kb + (size_t)(16 * j + r) * HD_;  // B column n = 16j+r
    acc[j] = wmma_bf16(qa0, b_frag_bf16(krow, 0, h), acc[j]);
    acc[j] = wmma_bf16(qa1, b_frag_bf16(krow, 32, h), acc[j]);
  }

  // scale + additive mask (before softmax, as in reference)
  const float scale = 0.125f;  // 1/sqrt(64)
  const float* mrow[8];
#pragma unroll
  for (int v = 0; v < 8; ++v) mrow[v] = mask + (size_t)(m0 + v + 8 * h) * T_ + r;
#pragma unroll
  for (int j = 0; j < 64; ++j)
#pragma unroll
    for (int v = 0; v < 8; ++v)
      acc[j][v] = acc[j][v] * scale + mrow[v][16 * j];

  // Row softmax. Row (m0 + v + 8h) lives in the 16 lanes of this half-wave
  // (cols r + 16j), so reduce over j then shfl_xor with masks 8,4,2,1.
  float inv[8];
#pragma unroll
  for (int v = 0; v < 8; ++v) {
    float mx = -3.0e38f;
#pragma unroll
    for (int j = 0; j < 64; ++j) mx = fmaxf(mx, acc[j][v]);
#pragma unroll
    for (int off = 8; off > 0; off >>= 1) mx = fmaxf(mx, __shfl_xor(mx, off, 32));
    float s = 0.0f;
#pragma unroll
    for (int j = 0; j < 64; ++j) {
      const float p = __expf(acc[j][v] - mx);
      acc[j][v] = p;
      s += p;
    }
#pragma unroll
    for (int off = 8; off > 0; off >>= 1) s += __shfl_xor(s, off, 32);
    inv[v] = 1.0f / s;
  }

  float* orow[8];
#pragma unroll
  for (int v = 0; v < 8; ++v)
    orow[v] = attn + ((size_t)bh * T_ + (m0 + v + 8 * h)) * T_ + r;
#pragma unroll
  for (int j = 0; j < 64; ++j)
#pragma unroll
    for (int v = 0; v < 8; ++v)
      orow[v][16 * j] = acc[j][v] * inv[v];
}

// ---- Kernel 3: ctx = attn @ V  (per head: 1024x64, K=1024), bf16 out --------
// attn read back as f32 (L2-resident: 192MB global L2), bf16-converted in the
// A-load. Vt rows are B columns (two 16B loads per lane).
__global__ __launch_bounds__(128, 1)
void attn_ctx_kernel(const float* __restrict__ attn, const bf16* __restrict__ Vt,
                     bf16* __restrict__ ctx) {
  const int lane = threadIdx.x & 31;
  const int wave = threadIdx.x >> 5;
  const int r = lane & 15, h = lane >> 4;
  const int bh = blockIdx.y;
  const int m0 = (blockIdx.x * 4 + wave) * 32;
  const float* arow0 = attn + ((size_t)bh * T_ + (m0 + r)) * T_;
  const float* arow1 = attn + ((size_t)bh * T_ + (m0 + 16 + r)) * T_;
  const bf16* vrow[4];
#pragma unroll
  for (int j = 0; j < 4; ++j)
    vrow[j] = Vt + ((size_t)bh * HD_ + 16 * j + r) * T_;

  v8f acc[2][4];
#pragma unroll
  for (int i = 0; i < 2; ++i)
#pragma unroll
    for (int j = 0; j < 4; ++j) acc[i][j] = vzero8();

  for (int k0 = 0; k0 < T_; k0 += 32) {
    const v16bf a0 = a_frag_f32(arow0, k0, h);
    const v16bf a1 = a_frag_f32(arow1, k0, h);
#pragma unroll
    for (int j = 0; j < 4; ++j) {
      const v16bf b = b_frag_bf16(vrow[j], k0, h);
      acc[0][j] = wmma_bf16(a0, b, acc[0][j]);
      acc[1][j] = wmma_bf16(a1, b, acc[1][j]);
    }
  }

  const int bb = bh >> 4, hh = bh & 15;
#pragma unroll
  for (int i = 0; i < 2; ++i)
#pragma unroll
    for (int j = 0; j < 4; ++j)
#pragma unroll
      for (int v = 0; v < 8; ++v) {
        const int t = m0 + 16 * i + v + 8 * h;
        ctx[((size_t)bb * T_ + t) * D_ + hh * HD_ + 16 * j + r] =
            (bf16)acc[i][j][v];
      }
}

// ---- Kernel 4: out = ctx @ Wo^T + bo, f32 into d_out (pre-LN) ---------------
__global__ __launch_bounds__(128, 1)
void out_proj_kernel(const bf16* __restrict__ Cx, const bf16* __restrict__ W,
                     const float* __restrict__ bias, float* __restrict__ out) {
  const int lane = threadIdx.x & 31;
  const int wave = threadIdx.x >> 5;
  const int r = lane & 15, h = lane >> 4;
  const int m0 = blockIdx.y * 128 + wave * 32;
  const int n0 = blockIdx.x * 64;
  const bf16* arow0 = Cx + (size_t)(m0 + r) * D_;
  const bf16* arow1 = Cx + (size_t)(m0 + 16 + r) * D_;
  const bf16* wrow[4];
#pragma unroll
  for (int j = 0; j < 4; ++j) wrow[j] = W + (size_t)(n0 + 16 * j + r) * D_;

  v8f acc[2][4];
#pragma unroll
  for (int i = 0; i < 2; ++i)
#pragma unroll
    for (int j = 0; j < 4; ++j) acc[i][j] = vzero8();

  for (int k0 = 0; k0 < D_; k0 += 32) {
    const v16bf a0 = a_frag_bf16(arow0, k0, h);
    const v16bf a1 = a_frag_bf16(arow1, k0, h);
#pragma unroll
    for (int j = 0; j < 4; ++j) {
      const v16bf b = b_frag_bf16(wrow[j], k0, h);
      acc[0][j] = wmma_bf16(a0, b, acc[0][j]);
      acc[1][j] = wmma_bf16(a1, b, acc[1][j]);
    }
  }

#pragma unroll
  for (int j = 0; j < 4; ++j) {
    const int n = n0 + 16 * j + r;
    const float bv = bias[n];
#pragma unroll
    for (int i = 0; i < 2; ++i)
#pragma unroll
      for (int v = 0; v < 8; ++v) {
        const int m = m0 + 16 * i + v + 8 * h;
        out[(size_t)m * D_ + n] = acc[i][j][v] + bv;
      }
  }
}

// ---- Kernel 5: residual + LayerNorm (in place on d_out) ---------------------
static __device__ __forceinline__ float block_reduce_sum(float v, float* sm) {
#pragma unroll
  for (int off = 16; off > 0; off >>= 1) v += __shfl_xor(v, off, 32);
  const int lane = threadIdx.x & 31, w = threadIdx.x >> 5;
  __syncthreads();
  if (lane == 0) sm[w] = v;
  __syncthreads();
  float t = sm[0];
#pragma unroll
  for (int i = 1; i < 8; ++i) t += sm[i];
  return t;
}

__global__ __launch_bounds__(256)
void residual_ln_kernel(float* __restrict__ out, const float* __restrict__ x,
                        const float* __restrict__ g, const float* __restrict__ b) {
  __shared__ float sm[8];
  const size_t row = blockIdx.x;
  const int tid = threadIdx.x;
  float vals[4];
  float s = 0.0f;
#pragma unroll
  for (int k = 0; k < 4; ++k) {
    const int c = tid + 256 * k;
    const float v = out[row * D_ + c] + x[row * D_ + c];
    vals[k] = v;
    s += v;
  }
  s = block_reduce_sum(s, sm);
  const float mu = s * (1.0f / D_);
  float vs = 0.0f;
#pragma unroll
  for (int k = 0; k < 4; ++k) { const float d = vals[k] - mu; vs += d * d; }
  vs = block_reduce_sum(vs, sm);
  const float rstd = rsqrtf(vs * (1.0f / D_) + EPS_);
#pragma unroll
  for (int k = 0; k < 4; ++k) {
    const int c = tid + 256 * k;
    out[row * D_ + c] = (vals[k] - mu) * rstd * g[c] + b[c];
  }
}

// ---- Host launcher ----------------------------------------------------------
extern "C" void kernel_launch(void* const* d_in, const int* in_sizes, int n_in,
                              void* d_out, int out_size, void* d_ws, size_t ws_size,
                              hipStream_t stream) {
  (void)in_sizes; (void)n_in; (void)out_size; (void)ws_size;
  const float* x    = (const float*)d_in[0];
  const float* mask = (const float*)d_in[1];
  const float* wq   = (const float*)d_in[2];
  const float* bq   = (const float*)d_in[3];
  const float* wk   = (const float*)d_in[4];
  const float* bk   = (const float*)d_in[5];
  const float* wv   = (const float*)d_in[6];
  const float* bv   = (const float*)d_in[7];
  const float* wo   = (const float*)d_in[8];
  const float* bo   = (const float*)d_in[9];
  const float* lnw  = (const float*)d_in[10];
  const float* lnb  = (const float*)d_in[11];

  float* out  = (float*)d_out;
  float* attn = out + (size_t)B_ * T_ * D_;  // outputs concatenated: (out, attn)

  // Workspace layout with liveness aliasing (42 MB total):
  //   [ 0,  8M): Xbf   (dead after projections)  --+ reused by ctx bf16
  //   [ 8, 10M): Wq_bf (dead after projections)    |
  //   [10, 12M): Wk_bf (dead after projections)    |
  //   [12, 14M): Wv_bf (dead after projections)  --+
  //   [16, 24M): Qh bf16 [B,H,T,HD]
  //   [24, 32M): Kh bf16 [B,H,T,HD]
  //   [32, 40M): Vt bf16 [B,H,HD,T]
  //   [40, 42M): Wo_bf
  char* ws = (char*)d_ws;
  const size_t MB = 1024 * 1024;
  bf16* Xbf = (bf16*)(ws);
  bf16* Wqb = (bf16*)(ws + 8 * MB);
  bf16* Wkb = (bf16*)(ws + 10 * MB);
  bf16* Wvb = (bf16*)(ws + 12 * MB);
  bf16* Qh  = (bf16*)(ws + 16 * MB);
  bf16* Kh  = (bf16*)(ws + 24 * MB);
  bf16* Vt  = (bf16*)(ws + 32 * MB);
  bf16* Wob = (bf16*)(ws + 40 * MB);
  bf16* ctx = (bf16*)(ws);  // aliases Xbf/Wq/Wk/Wv region (dead by then)

  const int nX4 = (B_ * T_ * D_) / 4;  // 1M vec4
  const int nW4 = (D_ * D_) / 4;       // 256K vec4
  f32_to_bf16_kernel<<<(nX4 + 255) / 256, 256, 0, stream>>>(x, Xbf, nX4);
  f32_to_bf16_kernel<<<(nW4 + 255) / 256, 256, 0, stream>>>(wq, Wqb, nW4);
  f32_to_bf16_kernel<<<(nW4 + 255) / 256, 256, 0, stream>>>(wk, Wkb, nW4);
  f32_to_bf16_kernel<<<(nW4 + 255) / 256, 256, 0, stream>>>(wv, Wvb, nW4);
  f32_to_bf16_kernel<<<(nW4 + 255) / 256, 256, 0, stream>>>(wo, Wob, nW4);

  const dim3 gGemm(16, 32);  // N/64, M/128 (M = B*T = 4096)
  qkv_proj_kernel<<<gGemm, 128, 0, stream>>>(Xbf, Wqb, bq, Qh, 0);
  qkv_proj_kernel<<<gGemm, 128, 0, stream>>>(Xbf, Wkb, bk, Kh, 0);
  qkv_proj_kernel<<<gGemm, 128, 0, stream>>>(Xbf, Wvb, bv, Vt, 1);

  attn_scores_softmax_kernel<<<dim3(16, 64), 128, 0, stream>>>(Qh, Kh, mask, attn);
  attn_ctx_kernel<<<dim3(8, 64), 128, 0, stream>>>(attn, Vt, ctx);

  out_proj_kernel<<<gGemm, 128, 0, stream>>>(ctx, Wob, bo, out);
  residual_ln_kernel<<<4096, 256, 0, stream>>>(out, x, lnw, lnb);
}
